// ResiDual_Block_20779051778116
// MI455X (gfx1250) — matrix operand order repaired and checked by reading, not compiled, not run on hardware
//
#include <hip/hip_runtime.h>
#include <hip/hip_bf16.h>

typedef __attribute__((ext_vector_type(16))) _Float16 v16h;
typedef __attribute__((ext_vector_type(8)))  float    v8f;

#define WMMA16(a, b, c) __builtin_amdgcn_wmma_f32_16x16x32_f16( \
    false, (a), false, (b), (short)0, (c), false, false)

static constexpr int Bv = 4, Sv = 1024, Dv = 1024, Hv = 16, DHv = 64;
static constexpr int Mv = Bv * Sv;  // 4096

// ---------------------------------------------------------------------------
// CDNA5 async global->LDS copy (16B per lane), tracked with ASYNCcnt.
// ---------------------------------------------------------------------------
__device__ __forceinline__ void async_ld_b128(void* lds_dst, const void* gsrc) {
  unsigned l = (unsigned)(size_t)lds_dst;  // low 32 bits of LDS aperture = LDS offset
  asm volatile("global_load_async_to_lds_b128 %0, %1, off"
               :: "v"(l), "v"(gsrc) : "memory");
}
__device__ __forceinline__ void wait_async0() {
  asm volatile("s_wait_asynccnt 0x0" ::: "memory");
}
__device__ __forceinline__ void wait_ds0() {
  asm volatile("s_wait_dscnt 0x0" ::: "memory");
}

// ---------------------------------------------------------------------------
// fp32 [K,N] -> f16 transposed [N,K] (tiled 32x32 LDS transpose)
// ---------------------------------------------------------------------------
__global__ __launch_bounds__(256) void cvt_transpose_kernel(
    const float* __restrict__ src, _Float16* __restrict__ dst) {
  __shared__ float tile[32][33];
  const int tid = threadIdx.x;
  const int tx = tid & 31, ty = tid >> 5;  // ty: 0..7
  const int nb = blockIdx.x * 32;          // N base (src col)
  const int kb = blockIdx.y * 32;          // K base (src row)
#pragma unroll
  for (int j = 0; j < 4; ++j)
    tile[ty + j * 8][tx] = src[(size_t)(kb + ty + j * 8) * Dv + nb + tx];
  __syncthreads();
#pragma unroll
  for (int j = 0; j < 4; ++j)
    dst[(size_t)(nb + ty + j * 8) * Dv + kb + tx] = (_Float16)tile[tx][ty + j * 8];
}

// ---------------------------------------------------------------------------
// LayerNorm over rows of D=1024: one block (256 thr) per row.
// ---------------------------------------------------------------------------
__global__ __launch_bounds__(256) void ln_kernel(const float* __restrict__ x,
                                                 const float* __restrict__ g,
                                                 const float* __restrict__ beta,
                                                 float* __restrict__ outf,
                                                 _Float16* __restrict__ outh) {
  __shared__ float red[16];
  const int row = blockIdx.x;
  const int tid = threadIdx.x;
  const float* xr = x + (size_t)row * Dv;

  float t[4];
  float s = 0.f, ss = 0.f;
#pragma unroll
  for (int j = 0; j < 4; ++j) {
    float v = xr[tid + j * 256];
    t[j] = v;
    s += v;
    ss += v * v;
  }
#pragma unroll
  for (int m = 16; m >= 1; m >>= 1) {
    s += __shfl_xor(s, m, 32);
    ss += __shfl_xor(ss, m, 32);
  }
  if ((tid & 31) == 0) {
    red[tid >> 5] = s;
    red[8 + (tid >> 5)] = ss;
  }
  __syncthreads();
  s = 0.f; ss = 0.f;
#pragma unroll
  for (int i = 0; i < 8; ++i) { s += red[i]; ss += red[8 + i]; }

  const float mean = s * (1.0f / Dv);
  const float var = ss * (1.0f / Dv) - mean * mean;
  const float inv = rsqrtf(var + 1e-5f);
#pragma unroll
  for (int j = 0; j < 4; ++j) {
    int c = tid + j * 256;
    float y = (t[j] - mean) * inv * g[c] + beta[c];
    if (outf) outf[(size_t)row * Dv + c] = y;
    if (outh) outh[(size_t)row * Dv + c] = (_Float16)y;
  }
}

// ---------------------------------------------------------------------------
// Dual-residual combine: y = LN(x_ln + o; gl,bl) + LN(x + o; gr,br)
// ---------------------------------------------------------------------------
__global__ __launch_bounds__(256) void combine_kernel(
    const float* __restrict__ x, const float* __restrict__ xln,
    const float* __restrict__ o, const float* __restrict__ gl,
    const float* __restrict__ bl, const float* __restrict__ gr,
    const float* __restrict__ br, float* __restrict__ yout,
    _Float16* __restrict__ youth) {
  __shared__ float red[32];
  const int row = blockIdx.x;
  const int tid = threadIdx.x;
  const float* xr = x + (size_t)row * Dv;
  const float* lr = xln + (size_t)row * Dv;
  const float* orow = o + (size_t)row * Dv;

  float a[4], c[4];
  float sa = 0.f, ssa = 0.f, sc = 0.f, ssc = 0.f;
#pragma unroll
  for (int j = 0; j < 4; ++j) {
    int idx = tid + j * 256;
    float ov = orow[idx];
    float av = lr[idx] + ov;
    float cv = xr[idx] + ov;
    a[j] = av; c[j] = cv;
    sa += av; ssa += av * av;
    sc += cv; ssc += cv * cv;
  }
#pragma unroll
  for (int m = 16; m >= 1; m >>= 1) {
    sa += __shfl_xor(sa, m, 32);
    ssa += __shfl_xor(ssa, m, 32);
    sc += __shfl_xor(sc, m, 32);
    ssc += __shfl_xor(ssc, m, 32);
  }
  if ((tid & 31) == 0) {
    int w = tid >> 5;
    red[w] = sa; red[8 + w] = ssa; red[16 + w] = sc; red[24 + w] = ssc;
  }
  __syncthreads();
  sa = ssa = sc = ssc = 0.f;
#pragma unroll
  for (int i = 0; i < 8; ++i) {
    sa += red[i]; ssa += red[8 + i]; sc += red[16 + i]; ssc += red[24 + i];
  }
  const float ma = sa * (1.0f / Dv);
  const float ia = rsqrtf(ssa * (1.0f / Dv) - ma * ma + 1e-5f);
  const float mc = sc * (1.0f / Dv);
  const float ic = rsqrtf(ssc * (1.0f / Dv) - mc * mc + 1e-5f);
#pragma unroll
  for (int j = 0; j < 4; ++j) {
    int col = tid + j * 256;
    float y = (a[j] - ma) * ia * gl[col] + bl[col] +
              (c[j] - mc) * ic * gr[col] + br[col];
    yout[(size_t)row * Dv + col] = y;
    youth[(size_t)row * Dv + col] = (_Float16)y;
  }
}

// ---------------------------------------------------------------------------
// Tiled WMMA GEMM: C[4096,1024] = A[4096,1024](f16) @ W(f16, stored [N,K])
// Block tile 128x64, 8 waves (32x32 each), K-chunks of 32, double-buffered
// async global->LDS staging. All fragment reads are contiguous b128 pairs.
// mode 0: Q/K -> f16 [B,H,S,DH];  mode 3: V -> f16 [B,H,DH,S]
// mode 1: FF1 -> +bias, erf-GELU, f16 row-major
// mode 2: FF2 -> +bias +resid, f32 row-major
// ---------------------------------------------------------------------------
__global__ __launch_bounds__(256) void gemm_kernel(
    const _Float16* __restrict__ A, const _Float16* __restrict__ Wt,
    const float* __restrict__ bias, const float* __restrict__ resid,
    float* __restrict__ outf, _Float16* __restrict__ outh, int mode) {
  __shared__ _Float16 As[2][128][40];  // 128x32, +8 pad (row = 80B, 16B mult)
  __shared__ _Float16 Bs[2][64][40];   // 64(N)x32(K), transposed weight tile

  const int tid = threadIdx.x;
  const int lane = tid & 31;
  const int w = tid >> 5;
  const int wm = w >> 1;       // 0..3
  const int wn = w & 1;        // 0..1
  const int m0 = blockIdx.y * 128;
  const int n0 = blockIdx.x * 64;
  const int halfsel = lane >> 4;
  const int l15 = lane & 15;

  // per-thread staging coordinates
  const int ar0 = tid >> 2, ac0 = (tid & 3) * 8;          // A chunk 0
  const int ar1 = (tid + 256) >> 2, ac1 = ac0;            // A chunk 1
  const int nr = tid >> 2, nc = (tid & 3) * 8;            // B chunk

  v8f acc[2][2];
  {
    v8f z = {};
#pragma unroll
    for (int ms = 0; ms < 2; ++ms)
#pragma unroll
      for (int ns = 0; ns < 2; ++ns) acc[ms][ns] = z;
  }

  // stage first K-slab into buffer 0
  async_ld_b128(&As[0][ar0][ac0], A + (size_t)(m0 + ar0) * 1024 + ac0);
  async_ld_b128(&As[0][ar1][ac1], A + (size_t)(m0 + ar1) * 1024 + ac1);
  async_ld_b128(&Bs[0][nr][nc], Wt + (size_t)(n0 + nr) * 1024 + nc);
  wait_async0();
  __syncthreads();

  for (int kb = 0; kb < 1024; kb += 32) {
    const int buf = (kb >> 5) & 1;
    if (kb + 32 < 1024) {  // prefetch next slab into the other buffer
      const int kn = kb + 32;
      async_ld_b128(&As[buf ^ 1][ar0][ac0], A + (size_t)(m0 + ar0) * 1024 + kn + ac0);
      async_ld_b128(&As[buf ^ 1][ar1][ac1], A + (size_t)(m0 + ar1) * 1024 + kn + ac1);
      async_ld_b128(&Bs[buf ^ 1][nr][nc], Wt + (size_t)(n0 + nr) * 1024 + kn + nc);
      __builtin_prefetch(A + (size_t)(m0 + ar0) * 1024 + kb + 64, 0, 1);
    }

    v16h afrag[2], bfrag[2];
#pragma unroll
    for (int ms = 0; ms < 2; ++ms) {
      const int rowA = wm * 32 + ms * 16 + l15;
      const int koff = halfsel * 16;
#pragma unroll
      for (int i = 0; i < 8; ++i) {
        afrag[ms][i] = As[buf][rowA][koff + i];
        afrag[ms][i + 8] = As[buf][rowA][koff + 8 + i];
      }
    }
#pragma unroll
    for (int ns = 0; ns < 2; ++ns) {
      const int colB = wn * 32 + ns * 16 + l15;
      const int kbase = halfsel * 16;
#pragma unroll
      for (int i = 0; i < 16; ++i) bfrag[ns][i] = Bs[buf][colB][kbase + i];
    }
#pragma unroll
    for (int ms = 0; ms < 2; ++ms)
#pragma unroll
      for (int ns = 0; ns < 2; ++ns)
        acc[ms][ns] = WMMA16(afrag[ms], bfrag[ns], acc[ms][ns]);

    wait_async0();
    __syncthreads();
  }

  // Epilogue (C layout: VGPR r -> row r (+8 upper half), N = lane&15)
#pragma unroll
  for (int ms = 0; ms < 2; ++ms)
#pragma unroll
    for (int ns = 0; ns < 2; ++ns)
#pragma unroll
      for (int r = 0; r < 8; ++r) {
        const int row = m0 + wm * 32 + ms * 16 + r + 8 * halfsel;
        const int col = n0 + wn * 32 + ns * 16 + l15;
        float v = acc[ms][ns][r];
        if (mode == 0) {  // Q/K -> [B,H,S,DH] f16
          int b = row >> 10, sidx = row & 1023;
          int hh = col >> 6, dh = col & 63;
          outh[(((size_t)(b * Hv + hh) * Sv + sidx) << 6) + dh] = (_Float16)v;
        } else if (mode == 3) {  // V -> [B,H,DH,S] f16
          int b = row >> 10, sidx = row & 1023;
          int hh = col >> 6, dh = col & 63;
          outh[(((size_t)(b * Hv + hh) * DHv + dh) << 10) + sidx] = (_Float16)v;
        } else if (mode == 1) {
          v += bias[col];
          v = v * 0.5f * (1.0f + erff(v * 0.70710678118f));
          outh[(size_t)row * 1024 + col] = (_Float16)v;
        } else {
          v += bias[col] + resid[(size_t)row * 1024 + col];
          outf[(size_t)row * 1024 + col] = v;
        }
      }
}

// ---------------------------------------------------------------------------
// Flash attention: grid (S/128, B*H), 8 waves, wave owns 16 q-rows.
// Q,K: f16 [B,H,S,DH]; V: f16 [B,H,DH,S] (so all fragment reads contiguous).
// Double-buffered async K/V staging. O: f32 [B,S,D].
// ---------------------------------------------------------------------------
__global__ __launch_bounds__(256) void attn_kernel(const _Float16* __restrict__ Q,
                                                   const _Float16* __restrict__ K,
                                                   const _Float16* __restrict__ Vt,
                                                   float* __restrict__ O) {
  __shared__ _Float16 Ks[2][64][72];   // [key][dh]
  __shared__ _Float16 Vs[2][64][72];   // [dh][key]
  __shared__ _Float16 Ps[8][16][72];   // per-wave P staging (C->A relayout)

  const int tid = threadIdx.x;
  const int lane = tid & 31;
  const int w = tid >> 5;
  const int bh = blockIdx.y;
  const int q0 = blockIdx.x * 128 + w * 16;
  const int halfsel = lane >> 4;
  const int l15 = lane & 15;

  const _Float16* Qb = Q + (size_t)bh * Sv * DHv;
  const _Float16* Kb = K + (size_t)bh * Sv * DHv;
  const _Float16* Vb = Vt + (size_t)bh * DHv * Sv;

  // staging coords: 64 rows x (64 or 128B): K rows are 128B (8 chunks of 16B)
  const int kr0 = tid >> 2, kc0 = (tid & 3) * 8;            // K: 64x64 halves -> 512 chunks
  const int vr0 = tid >> 3, vc0 = (tid & 7) * 8;            // V: 64x64 halves... (same shape)

  // Q fragments (resident): 2 K-chunks over DH=64
  v16h qf[2];
  {
    const int rowQ = q0 + l15;
    const int koff = halfsel * 16;
#pragma unroll
    for (int kc = 0; kc < 2; ++kc)
#pragma unroll
      for (int i = 0; i < 8; ++i) {
        qf[kc][i] = Qb[(size_t)rowQ * DHv + kc * 32 + koff + i];
        qf[kc][i + 8] = Qb[(size_t)rowQ * DHv + kc * 32 + koff + 8 + i];
      }
  }

  v8f oacc[4];
  {
    v8f z = {};
#pragma unroll
    for (int t = 0; t < 4; ++t) oacc[t] = z;
  }
  float m_i[8], l_i[8];
#pragma unroll
  for (int r = 0; r < 8; ++r) { m_i[r] = -1e30f; l_i[r] = 0.f; }

  // stage first K/V chunk into buffer 0 (each tile: 64x64 halves = 512 chunks)
#pragma unroll
  for (int i = 0; i < 2; ++i) {
    int cch = tid + i * 256;
    int rr = cch >> 3, cc = (cch & 7) * 8;
    async_ld_b128(&Ks[0][rr][cc], Kb + (size_t)rr * DHv + cc);
    async_ld_b128(&Vs[0][rr][cc], Vb + (size_t)rr * Sv + cc);
  }
  wait_async0();
  __syncthreads();

  for (int key0 = 0; key0 < Sv; key0 += 64) {
    const int buf = (key0 >> 6) & 1;
    if (key0 + 64 < Sv) {
      const int kn = key0 + 64;
#pragma unroll
      for (int i = 0; i < 2; ++i) {
        int cch = tid + i * 256;
        int rr = cch >> 3, cc = (cch & 7) * 8;
        async_ld_b128(&Ks[buf ^ 1][rr][cc], Kb + (size_t)(kn + rr) * DHv + cc);
        async_ld_b128(&Vs[buf ^ 1][rr][cc], Vb + (size_t)rr * Sv + kn + cc);
      }
    }

    // S = Q @ K^T for 4 key sub-tiles of 16 (B column = contiguous K row)
    v8f sc[4];
    {
      v8f z = {};
#pragma unroll
      for (int t = 0; t < 4; ++t) sc[t] = z;
    }
#pragma unroll
    for (int t = 0; t < 4; ++t) {
#pragma unroll
      for (int kc = 0; kc < 2; ++kc) {
        v16h bf;
        const int colK = t * 16 + l15;
        const int kbase = kc * 32 + halfsel * 16;
#pragma unroll
        for (int i = 0; i < 16; ++i) bf[i] = Ks[buf][colK][kbase + i];
        sc[t] = WMMA16(qf[kc], bf, sc[t]);
      }
    }

    // Online softmax (masks 8..1 stay within each 16-lane C half)
#pragma unroll
    for (int r = 0; r < 8; ++r) {
      float mx = -1e30f;
#pragma unroll
      for (int t = 0; t < 4; ++t) {
        sc[t][r] *= 0.125f;
        mx = fmaxf(mx, sc[t][r]);
      }
#pragma unroll
      for (int msk = 8; msk >= 1; msk >>= 1) mx = fmaxf(mx, __shfl_xor(mx, msk, 32));
      const float mn = fmaxf(m_i[r], mx);
      const float scale = __expf(m_i[r] - mn);
      float ps = 0.f;
#pragma unroll
      for (int t = 0; t < 4; ++t) {
        float p = __expf(sc[t][r] - mn);
        sc[t][r] = p;
        ps += p;
      }
#pragma unroll
      for (int msk = 8; msk >= 1; msk >>= 1) ps += __shfl_xor(ps, msk, 32);
      l_i[r] = l_i[r] * scale + ps;
      m_i[r] = mn;
#pragma unroll
      for (int t = 0; t < 4; ++t) oacc[t][r] *= scale;
    }

    // P (C-layout) -> A-fragment layout via per-wave LDS
#pragma unroll
    for (int t = 0; t < 4; ++t)
#pragma unroll
      for (int r = 0; r < 8; ++r)
        Ps[w][r + 8 * halfsel][t * 16 + l15] = (_Float16)sc[t][r];
    wait_ds0();

    // O += P @ V  (V fragment: contiguous key run at fixed dh)
#pragma unroll
    for (int kc = 0; kc < 2; ++kc) {
      v16h pf;
      {
        const int koff = halfsel * 16;
#pragma unroll
        for (int i = 0; i < 8; ++i) {
          pf[i] = Ps[w][l15][kc * 32 + koff + i];
          pf[i + 8] = Ps[w][l15][kc * 32 + koff + 8 + i];
        }
      }
#pragma unroll
      for (int t = 0; t < 4; ++t) {
        v16h vf;
        const int colV = t * 16 + l15;             // dh
        const int kbase = kc * 32 + halfsel * 16;  // key local
#pragma unroll
        for (int i = 0; i < 16; ++i) vf[i] = Vs[buf][colV][kbase + i];
        oacc[t] = WMMA16(pf, vf, oacc[t]);
      }
    }

    wait_async0();
    __syncthreads();
  }

  // Normalize and scatter O to [B,S,D] f32
  const int b = bh >> 4;
  const int hh = bh & 15;
#pragma unroll
  for (int t = 0; t < 4; ++t)
#pragma unroll
    for (int r = 0; r < 8; ++r) {
      const int qrow = q0 + r + 8 * halfsel;
      const int col = hh * DHv + t * 16 + l15;
      O[((size_t)b * Sv + qrow) * Dv + col] = oacc[t][r] / l_i[r];
    }
}

// ---------------------------------------------------------------------------
extern "C" void kernel_launch(void* const* d_in, const int* in_sizes, int n_in,
                              void* d_out, int out_size, void* d_ws, size_t ws_size,
                              hipStream_t stream) {
  const float* x  = (const float*)d_in[0];
  const float* Wq = (const float*)d_in[1];
  const float* Wk = (const float*)d_in[2];
  const float* Wv = (const float*)d_in[3];
  const float* g0 = (const float*)d_in[4];
  const float* b0 = (const float*)d_in[5];
  const float* gl = (const float*)d_in[6];
  const float* bl = (const float*)d_in[7];
  const float* gr = (const float*)d_in[8];
  const float* br = (const float*)d_in[9];
  const float* gf = (const float*)d_in[10];
  const float* bf = (const float*)d_in[11];
  const float* W1 = (const float*)d_in[12];
  const float* b1 = (const float*)d_in[13];
  const float* W2 = (const float*)d_in[14];
  const float* b2 = (const float*)d_in[15];
  float* out = (float*)d_out;

  char* ws = (char*)d_ws;
  const size_t MiB = 1u << 20;
  float*    x_ln  = (float*)(ws + 0 * MiB);      // 16 MiB
  float*    ybuf  = (float*)(ws + 16 * MiB);     // 16 MiB
  float*    y2    = (float*)(ws + 32 * MiB);     // 16 MiB
  float*    obuf  = (float*)(ws + 48 * MiB);     // 16 MiB
  _Float16* xln_h = (_Float16*)(ws + 64 * MiB);  // 8 MiB
  _Float16* y_h   = (_Float16*)(ws + 72 * MiB);  // 8 MiB
  _Float16* h_h   = (_Float16*)(ws + 80 * MiB);  // 8 MiB
  _Float16* q_h   = (_Float16*)(ws + 88 * MiB);  // 8 MiB
  _Float16* k_h   = (_Float16*)(ws + 96 * MiB);  // 8 MiB
  _Float16* v_h   = (_Float16*)(ws + 104 * MiB); // 8 MiB
  _Float16* Wq_t  = (_Float16*)(ws + 112 * MiB); // 5 x 2 MiB (transposed f16)
  _Float16* Wk_t  = Wq_t + (1 << 20);
  _Float16* Wv_t  = Wk_t + (1 << 20);
  _Float16* W1_t  = Wv_t + (1 << 20);
  _Float16* W2_t  = W1_t + (1 << 20);

  const dim3 blk(256);
  const dim3 tgrid(Dv / 32, Dv / 32);  // (32, 32)
  cvt_transpose_kernel<<<tgrid, blk, 0, stream>>>(Wq, Wq_t);
  cvt_transpose_kernel<<<tgrid, blk, 0, stream>>>(Wk, Wk_t);
  cvt_transpose_kernel<<<tgrid, blk, 0, stream>>>(Wv, Wv_t);
  cvt_transpose_kernel<<<tgrid, blk, 0, stream>>>(W1, W1_t);
  cvt_transpose_kernel<<<tgrid, blk, 0, stream>>>(W2, W2_t);

  ln_kernel<<<Mv, blk, 0, stream>>>(x, g0, b0, x_ln, xln_h);

  const dim3 ggrid(Dv / 64, Mv / 128);  // (16, 32)
  gemm_kernel<<<ggrid, blk, 0, stream>>>(xln_h, Wq_t, nullptr, nullptr, nullptr, q_h, 0);
  gemm_kernel<<<ggrid, blk, 0, stream>>>(xln_h, Wk_t, nullptr, nullptr, nullptr, k_h, 0);
  gemm_kernel<<<ggrid, blk, 0, stream>>>(xln_h, Wv_t, nullptr, nullptr, nullptr, v_h, 3);

  const dim3 agrid(Sv / 128, Bv * Hv);  // (8, 64)
  attn_kernel<<<agrid, blk, 0, stream>>>(q_h, k_h, v_h, obuf);

  combine_kernel<<<Mv, blk, 0, stream>>>(x, x_ln, obuf, gl, bl, gr, br, ybuf, y_h);

  gemm_kernel<<<ggrid, blk, 0, stream>>>(y_h, W1_t, b1, nullptr, nullptr, h_h, 1);
  gemm_kernel<<<ggrid, blk, 0, stream>>>(h_h, W2_t, b2, ybuf, y2, nullptr, 2);

  ln_kernel<<<Mv, blk, 0, stream>>>(y2, gf, bf, out, nullptr);
}
